// Attention_86981677679333
// MI455X (gfx1250) — matrix operand order repaired and checked by reading, not compiled, not run on hardware
//
#include <hip/hip_runtime.h>
#include <hip/hip_bf16.h>

// ---------------------------------------------------------------------------
// MI455X (gfx1250) attention with QK-LayerNorm.
// All matmuls go through v_wmma_f32_16x16x32_f16 (wave32 WMMA), fp32 accum.
// All LDS fragment loads are contiguous 16/32B so they lower to ds_load_b128.
// K tiles in the attention kernel are moved by the Tensor Data Mover
// (tensor_load_to_lds + s_wait_tensorcnt), with TDM padding matching the
// padded LDS layout.
// ---------------------------------------------------------------------------

typedef __attribute__((ext_vector_type(16))) _Float16 v16h;
typedef __attribute__((ext_vector_type(8)))  _Float16 h8;
typedef __attribute__((ext_vector_type(8)))  float    v8f;
typedef __attribute__((ext_vector_type(4)))  unsigned int v4u;
typedef __attribute__((ext_vector_type(8)))  int v8i;
typedef __attribute__((ext_vector_type(4)))  int v4i;

union Frag { v16h h; unsigned int u[8]; };

#define DIMC   1024
#define NHEAD  16
#define HD     64
#define SEQ    2048
#define NBATCH 2
#define MROWS  (NBATCH * SEQ)     /* 4096 */

#if __has_builtin(__builtin_amdgcn_tensor_load_to_lds) && \
    __has_builtin(__builtin_amdgcn_s_wait_tensorcnt)
#define USE_TDM 1
#endif

// ----------------------------- workspace layout ----------------------------
// all offsets 256B aligned; total ~96 MB
#define OFF_XB     ((size_t)0)                       /* f16 [4096][1024]  8MB */
#define OFF_WQKV   ((size_t)8388608)                 /* f16 [1024][3072]  6MB */
#define OFF_WPROJ  ((size_t)14680064)                /* f16 [1024][1024]  2MB */
#define OFF_QKV    ((size_t)16777216)                /* f32 [4096][3072] 48MB */
#define OFF_QB     ((size_t)67108864)                /* f16 [B,H,N,64]    8MB */
#define OFF_KB     ((size_t)75497472)                /* f16 [B,H,N,64]    8MB */
#define OFF_VB     ((size_t)83886080)                /* f16 [B,H,N,64]    8MB */
#define OFF_ATTN   ((size_t)92274688)                /* f16 [4096][1024]  8MB */

#ifdef USE_TDM
// ---------------------------------------------------------------------------
// TDM 2D tile load: rows x rowlen f16 elements, global row stride
// stride_elems, into LDS at lds_addr with optional LDS padding
// (pad codes per D# group1: interval 4 => 32 dwords, amount 3 => 4 dwords).
// Descriptor layout per CDNA5 ISA.
// ---------------------------------------------------------------------------
__device__ __forceinline__ void tdm_load_tile_f16(
    unsigned lds_addr, const void* gptr, unsigned rows, unsigned rowlen,
    unsigned stride_elems, unsigned pad_interval_code, unsigned pad_amount_code,
    bool pad_en) {
  const unsigned long long ga = (unsigned long long)gptr;
  v4u g0 = {0u, 0u, 0u, 0u};
  g0[0] = 1u;                                     // count=1, user descriptor
  g0[1] = lds_addr;                               // lds_addr (bytes)
  g0[2] = (unsigned)(ga & 0xffffffffu);           // global_addr[31:0]
  g0[3] = (unsigned)((ga >> 32) & 0x01ffffffu)    // global_addr[56:32]
        | (2u << 30);                             // type = 2 ("image")
  v8i g1 = {0, 0, 0, 0, 0, 0, 0, 0};
  g1[0] = (int)((1u << 16)                        // data_size = 2 bytes
        | (pad_en ? (1u << 20) : 0u)              // pad_enable
        | (pad_interval_code << 22)               // pad_interval
        | (pad_amount_code << 25));               // pad_amount
  g1[1] = (int)((rowlen & 0xffffu) << 16);        // tensor_dim0[15:0]
  g1[2] = (int)(((rowlen >> 16) & 0xffffu)        // tensor_dim0[31:16]
        | ((rows & 0xffffu) << 16));              // tensor_dim1[15:0]
  g1[3] = (int)(((rows >> 16) & 0xffffu)          // tensor_dim1[31:16]
        | ((rowlen & 0xffffu) << 16));            // tile_dim0
  g1[4] = (int)(rows & 0xffffu);                  // tile_dim1 (tile_dim2 = 0)
  g1[5] = (int)stride_elems;                      // tensor_dim0_stride[31:0]
  v4i g2 = {0, 0, 0, 0};                          // <=2D tensor: groups 2/3 zero
  v4i g3 = {0, 0, 0, 0};
#if __clang_major__ >= 23
  v8i g4 = {0, 0, 0, 0, 0, 0, 0, 0};
  __builtin_amdgcn_tensor_load_to_lds(g0, g1, g2, g3, g4, 0);   // 6-arg form
#else
  __builtin_amdgcn_tensor_load_to_lds(g0, g1, g2, g3, 0);       // 5-arg form
#endif
}
#endif  // USE_TDM

// --------------------------- fp32 -> f16 convert ---------------------------
__global__ void cvt_f32_f16(const float* __restrict__ in,
                            _Float16* __restrict__ out, int n) {
  int i = blockIdx.x * blockDim.x + threadIdx.x;
  int stride = gridDim.x * blockDim.x;
  for (; i < n; i += stride) out[i] = (_Float16)in[i];
}

// -------- per-head LayerNorm over hd=64 (1 wave / row), f32 in, f16 out ----
__global__ void qk_layernorm(const float* __restrict__ qkv, int col_off,
                             const float* __restrict__ scale,
                             const float* __restrict__ bias,
                             _Float16* __restrict__ out) {
  const int wave = threadIdx.x >> 5;
  const int lane = threadIdx.x & 31;
  const int row  = blockIdx.x * 8 + wave;               // (b*H + h)*N + n
  const int b = row >> 15;
  const int h = (row >> 11) & (NHEAD - 1);
  const int n = row & (SEQ - 1);
  const float* src = qkv + (size_t)(b * SEQ + n) * (3 * DIMC) + col_off + h * HD;
  float x0 = src[lane * 2 + 0];
  float x1 = src[lane * 2 + 1];
  float s = x0 + x1;
  for (int o = 1; o < 32; o <<= 1) s += __shfl_xor(s, o, 32);
  const float mean = s * (1.0f / 64.0f);
  const float d0 = x0 - mean, d1 = x1 - mean;
  float v = d0 * d0 + d1 * d1;
  for (int o = 1; o < 32; o <<= 1) v += __shfl_xor(v, o, 32);
  const float rstd = rsqrtf(v * (1.0f / 64.0f) + 1e-6f);
  _Float16* dst = out + (size_t)row * HD;
  dst[lane * 2 + 0] = (_Float16)(d0 * rstd * scale[lane * 2 + 0] + bias[lane * 2 + 0]);
  dst[lane * 2 + 1] = (_Float16)(d1 * rstd * scale[lane * 2 + 1] + bias[lane * 2 + 1]);
}

// ------------- gather V out of fused QKV (f32) into [B,H,N,64] f16 ---------
__global__ void v_gather(const float* __restrict__ qkv, _Float16* __restrict__ out,
                         int total) {
  int i = blockIdx.x * blockDim.x + threadIdx.x;
  if (i >= total) return;
  const int row = i >> 6;           // (b*H + h)*N + n
  const int d = i & 63;
  const int b = row >> 15;
  const int h = (row >> 11) & (NHEAD - 1);
  const int n = row & (SEQ - 1);
  out[i] = (_Float16)qkv[(size_t)(b * SEQ + n) * (3 * DIMC) + 2 * DIMC + h * HD + d];
}

// ---------------------------------------------------------------------------
// Generic f16 WMMA GEMM: C[M][N] = A[M][K] @ B[K][N] + bias,   fp32 output.
// Tile 128x128x32, 256 threads = 8 waves (4 M x 2 N), each wave 32x64 =
// 2x4 grid of 16x16 fp32 accumulators -> 8 v_wmma per K-step.
// B tile is stored in LDS transposed/pair-packed (Bp[col][k]) so every
// fragment load is a contiguous 32B -> 2x ds_load_b128.
// Register-staged double buffering: 1 barrier per K-step.
// ---------------------------------------------------------------------------
__global__ void __launch_bounds__(256)
gemm_f16_wmma(const _Float16* __restrict__ A, const _Float16* __restrict__ B,
              const float* __restrict__ bias, float* __restrict__ C,
              int M, int N, int K) {
  __shared__ _Float16 As[2][128][40];   // [M][K], row stride 80B (16B multiple)
  __shared__ _Float16 Bp[2][128][40];   // [N][K] transposed, stride 80B

  const int tid  = threadIdx.x;
  const int lane = tid & 31;
  const int wave = tid >> 5;
  const int wm = wave >> 1, wn = wave & 1;
  const int m0 = blockIdx.y * 128;
  const int n0 = blockIdx.x * 128;
  const int lhalf = lane >> 4;          // WMMA lane-half selector
  const int l16   = lane & 15;

  v8f acc[2][4];
  for (int i = 0; i < 2; ++i)
    for (int j = 0; j < 4; ++j)
      for (int r = 0; r < 8; ++r) acc[i][j][r] = 0.0f;

  // prologue: stage tile 0 into buffer 0
  {
    for (int it = 0; it < 2; ++it) {
      const int c = tid + it * 256;
      const int ar = c >> 2, ac = (c & 3) * 8;
      *(h8*)&As[0][ar][ac] = *(const h8*)&A[(size_t)(m0 + ar) * K + ac];
      const int br = c >> 4, bc = (c & 15) * 8;
      const h8 bv = *(const h8*)&B[(size_t)br * N + n0 + bc];
      for (int i = 0; i < 8; ++i) Bp[0][bc + i][br] = bv[i];
    }
  }
  __syncthreads();

  for (int k0 = 0; k0 < K; k0 += 32) {
    const int cur = (k0 >> 5) & 1;
    const int nxt = cur ^ 1;
    const bool more = (k0 + 32) < K;

    // fetch next tile from global into registers (overlaps with WMMA below)
    h8 aregs[2], bregs[2];
    if (more) {
      for (int it = 0; it < 2; ++it) {
        const int c = tid + it * 256;
        const int ar = c >> 2, ac = (c & 3) * 8;
        aregs[it] = *(const h8*)&A[(size_t)(m0 + ar) * K + (k0 + 32) + ac];
        const int br = c >> 4, bc = (c & 15) * 8;
        bregs[it] = *(const h8*)&B[(size_t)(k0 + 32 + br) * N + n0 + bc];
      }
      if (k0 + 64 < K) {  // global_prefetch_b8 for the tile 2 steps ahead
        __builtin_prefetch(&A[(size_t)(m0 + (tid >> 1)) * K + k0 + 64], 0, 1);
        __builtin_prefetch(&B[(size_t)(k0 + 64 + (tid >> 3)) * N + n0 + (tid & 7) * 16], 0, 1);
      }
    }

    // compute from LDS[cur]: all fragment reads contiguous -> ds_load_b128
    Frag af[2];
    for (int mt = 0; mt < 2; ++mt) {
      const int row = wm * 32 + mt * 16 + l16;
      __builtin_memcpy(&af[mt].u[0], &As[cur][row][8 * lhalf], 16);
      __builtin_memcpy(&af[mt].u[4], &As[cur][row][16 + 8 * lhalf], 16);
    }
    for (int nt = 0; nt < 4; ++nt) {
      Frag bf;
      const int col = wn * 64 + nt * 16 + l16;
      __builtin_memcpy(&bf.u[0], &Bp[cur][col][8 * lhalf], 16);
      __builtin_memcpy(&bf.u[4], &Bp[cur][col][16 + 8 * lhalf], 16);
      for (int mt = 0; mt < 2; ++mt)
        acc[mt][nt] = __builtin_amdgcn_wmma_f32_16x16x32_f16(
            false, af[mt].h, false, bf.h, (short)0, acc[mt][nt], false, false);
    }

    // store staged registers into the alternate LDS buffer
    if (more) {
      for (int it = 0; it < 2; ++it) {
        const int c = tid + it * 256;
        const int ar = c >> 2, ac = (c & 3) * 8;
        *(h8*)&As[nxt][ar][ac] = aregs[it];
        const int br = c >> 4, bc = (c & 15) * 8;
        for (int i = 0; i < 8; ++i) Bp[nxt][bc + i][br] = bregs[it][i];
      }
    }
    __syncthreads();
  }

  // epilogue: C-layout rows m = r + 8*lhalf, cols = l16
  for (int nt = 0; nt < 4; ++nt) {
    const int col = n0 + wn * 64 + nt * 16 + l16;
    const float bv = bias ? bias[col] : 0.0f;
    for (int mt = 0; mt < 2; ++mt)
      for (int r = 0; r < 8; ++r) {
        const int row = m0 + wm * 32 + mt * 16 + r + 8 * lhalf;
        C[(size_t)row * N + col] = acc[mt][nt][r] + bv;
      }
  }
}

// ---------------------------------------------------------------------------
// Flash attention per (b, h, 64-query block). 128 threads = 4 waves; each
// wave owns 16 query rows. Per 32-key step: 4 WMMA for S = Q K^T, online
// softmax in fp32 (16-lane xor reductions match C-layout lane halves),
// P re-swizzled C-layout -> A-layout through per-wave LDS, 4 WMMA for P V.
// K tiles move via the Tensor Data Mover (wave 0 issues, waits TENSORcnt);
// TDM LDS padding (32 dwords data + 4 dwords pad) reproduces Ks[32][72].
// V is staged transposed (Vt[d][key]) so PV B-fragments are contiguous 32B.
// ---------------------------------------------------------------------------
__global__ void __launch_bounds__(128)
flash_attn_wmma(const _Float16* __restrict__ Q, const _Float16* __restrict__ Kt,
                const _Float16* __restrict__ V, _Float16* __restrict__ O) {
  __shared__ _Float16 Ks[32][72];       // 32 keys x 64 dims, stride 144B
  __shared__ _Float16 Vt[64][40];       // transposed: [d][key], stride 80B
  __shared__ _Float16 Ps[4][16][40];    // per-wave P scratch, stride 80B

  const int tid  = threadIdx.x;
  const int lane = tid & 31;
  const int wave = tid >> 5;
  const int lhalf = lane >> 4;
  const int l16   = lane & 15;
  const int h = blockIdx.y, b = blockIdx.z;
  const int bh = b * NHEAD + h;
  const int q0 = blockIdx.x * 64 + wave * 16;
  const size_t base = (size_t)bh * SEQ * HD;

  // resident Q A-fragments (16 rows x 64 dims -> 2 fragments of K=32)
  Frag qf[2];
  {
    const _Float16* qsrc = Q + base + (size_t)(q0 + l16) * HD;
    for (int f = 0; f < 2; ++f) {
      __builtin_memcpy(&qf[f].u[0], &qsrc[f * 32 + 8 * lhalf], 16);
      __builtin_memcpy(&qf[f].u[4], &qsrc[f * 32 + 16 + 8 * lhalf], 16);
    }
  }

  float mrow[8], lrow[8];
  v8f oacc[4];
  for (int r = 0; r < 8; ++r) { mrow[r] = -1e30f; lrow[r] = 0.0f; }
  for (int dt = 0; dt < 4; ++dt)
    for (int r = 0; r < 8; ++r) oacc[dt][r] = 0.0f;

  const float scale = 0.125f;  // 1/sqrt(64)

  for (int kv0 = 0; kv0 < SEQ; kv0 += 32) {
#ifdef USE_TDM
    // K tile via Tensor Data Mover: 32 rows x 64 f16, padded into Ks[32][72]
    if (wave == 0) {
      tdm_load_tile_f16((unsigned)(size_t)&Ks[0][0],
                        &Kt[base + (size_t)kv0 * HD],
                        /*rows=*/32, /*rowlen=*/64, /*stride=*/64,
                        /*pad_interval: 32 dwords=*/4, /*pad_amount: 4 dwords=*/3,
                        /*pad_en=*/true);
      __builtin_amdgcn_s_wait_tensorcnt(0);
    }
    // V transposed staging by all 128 threads
    for (int it = 0; it < 2; ++it) {
      const int c = tid + it * 128;
      const int r = c >> 3, cc = (c & 7) * 8;
      const h8 vv = *(const h8*)&V[base + (size_t)(kv0 + r) * HD + cc];
      for (int i = 0; i < 8; ++i) Vt[cc + i][r] = vv[i];
    }
#else
    // cooperative K/V tile load (32x64 each); K row-major, V transposed
    for (int it = 0; it < 2; ++it) {
      const int c = tid + it * 128;
      const int r = c >> 3, cc = (c & 7) * 8;
      *(h8*)&Ks[r][cc] = *(const h8*)&Kt[base + (size_t)(kv0 + r) * HD + cc];
      const h8 vv = *(const h8*)&V[base + (size_t)(kv0 + r) * HD + cc];
      for (int i = 0; i < 8; ++i) Vt[cc + i][r] = vv[i];
    }
#endif
    if (kv0 + 32 < SEQ) {
      __builtin_prefetch(&Kt[base + (size_t)(kv0 + 32 + (tid >> 2)) * HD], 0, 1);
      __builtin_prefetch(&V [base + (size_t)(kv0 + 32 + (tid >> 2)) * HD], 0, 1);
    }
    __syncthreads();

    // S = Q K^T : two 16x16 key tiles, K-dim = hd = 64 -> 2 chained WMMAs
    v8f sacc[2];
    for (int nt = 0; nt < 2; ++nt) {
      for (int r = 0; r < 8; ++r) sacc[nt][r] = 0.0f;
      const int key = nt * 16 + l16;
      for (int f = 0; f < 2; ++f) {
        Frag kf;   // B[d][key] = Ks[key][d]; two 16B chunks along d (table order)
        __builtin_memcpy(&kf.u[0], &Ks[key][f * 32 + 8 * lhalf], 16);
        __builtin_memcpy(&kf.u[4], &Ks[key][f * 32 + 16 + 8 * lhalf], 16);
        sacc[nt] = __builtin_amdgcn_wmma_f32_16x16x32_f16(
            false, qf[f].h, false, kf.h, (short)0, sacc[nt], false, false);
      }
    }

    // online softmax per query row (row m = r + 8*lhalf)
    for (int r = 0; r < 8; ++r) {
      float s0 = sacc[0][r] * scale, s1 = sacc[1][r] * scale;
      float mx = fmaxf(s0, s1);
      for (int o = 1; o < 16; o <<= 1) mx = fmaxf(mx, __shfl_xor(mx, o, 32));
      const float mnew  = fmaxf(mrow[r], mx);
      const float alpha = __expf(mrow[r] - mnew);
      const float p0 = __expf(s0 - mnew), p1 = __expf(s1 - mnew);
      float ps = p0 + p1;
      for (int o = 1; o < 16; o <<= 1) ps += __shfl_xor(ps, o, 32);
      lrow[r] = lrow[r] * alpha + ps;
      mrow[r] = mnew;
      for (int dt = 0; dt < 4; ++dt) oacc[dt][r] *= alpha;
      const int m = r + 8 * lhalf;
      Ps[wave][m][l16]      = (_Float16)p0;
      Ps[wave][m][16 + l16] = (_Float16)p1;
    }
    // per-wave C-layout -> A-layout round trip through LDS; order the ds ops
    asm volatile("s_wait_dscnt 0" ::: "memory");

    Frag pf;   // P as 16x32 A-fragment (K = 32 keys)
    __builtin_memcpy(&pf.u[0], &Ps[wave][l16][8 * lhalf], 16);
    __builtin_memcpy(&pf.u[4], &Ps[wave][l16][16 + 8 * lhalf], 16);
    for (int dt = 0; dt < 4; ++dt) {
      Frag vf;   // B[key][d]: two 16B chunks along key in Vt (table order)
      const int dcol = dt * 16 + l16;
      __builtin_memcpy(&vf.u[0], &Vt[dcol][8 * lhalf], 16);
      __builtin_memcpy(&vf.u[4], &Vt[dcol][16 + 8 * lhalf], 16);
      oacc[dt] = __builtin_amdgcn_wmma_f32_16x16x32_f16(
          false, pf.h, false, vf.h, (short)0, oacc[dt], false, false);
    }
    __syncthreads();
  }

  // normalize and write in [b, n, h*64+d] layout (proj GEMM input)
  for (int dt = 0; dt < 4; ++dt)
    for (int r = 0; r < 8; ++r) {
      const int m = r + 8 * lhalf;
      const float val = oacc[dt][r] / lrow[r];
      O[(size_t)(b * SEQ + q0 + m) * DIMC + h * HD + dt * 16 + l16] = (_Float16)val;
    }
}

// ---------------------------------------------------------------------------
extern "C" void kernel_launch(void* const* d_in, const int* in_sizes, int n_in,
                              void* d_out, int out_size, void* d_ws, size_t ws_size,
                              hipStream_t stream) {
  const float* x      = (const float*)d_in[0];
  const float* qkv_w  = (const float*)d_in[1];
  const float* qkv_b  = (const float*)d_in[2];
  const float* qn_s   = (const float*)d_in[3];
  const float* qn_b   = (const float*)d_in[4];
  const float* kn_s   = (const float*)d_in[5];
  const float* kn_b   = (const float*)d_in[6];
  const float* proj_w = (const float*)d_in[7];
  const float* proj_b = (const float*)d_in[8];
  float* out = (float*)d_out;

  char* ws = (char*)d_ws;
  _Float16* xb    = (_Float16*)(ws + OFF_XB);
  _Float16* wqkv  = (_Float16*)(ws + OFF_WQKV);
  _Float16* wproj = (_Float16*)(ws + OFF_WPROJ);
  float*    qkvf  = (float*)   (ws + OFF_QKV);
  _Float16* qb    = (_Float16*)(ws + OFF_QB);
  _Float16* kb    = (_Float16*)(ws + OFF_KB);
  _Float16* vb    = (_Float16*)(ws + OFF_VB);
  _Float16* attn  = (_Float16*)(ws + OFF_ATTN);

  // 1) precision conversion of activations + weights (bandwidth-trivial)
  cvt_f32_f16<<<1024, 256, 0, stream>>>(x,      xb,    MROWS * DIMC);
  cvt_f32_f16<<<1024, 256, 0, stream>>>(qkv_w,  wqkv,  DIMC * 3 * DIMC);
  cvt_f32_f16<<<512,  256, 0, stream>>>(proj_w, wproj, DIMC * DIMC);

  // 2) fused QKV GEMM: [4096,1024] x [1024,3072] -> fp32
  gemm_f16_wmma<<<dim3(3 * DIMC / 128, MROWS / 128), 256, 0, stream>>>(
      xb, wqkv, qkv_b, qkvf, MROWS, 3 * DIMC, DIMC);

  // 3) per-head LayerNorm on q,k (fp32 stats), gather-convert v
  qk_layernorm<<<NBATCH * NHEAD * SEQ / 8, 256, 0, stream>>>(qkvf, 0,    qn_s, qn_b, qb);
  qk_layernorm<<<NBATCH * NHEAD * SEQ / 8, 256, 0, stream>>>(qkvf, DIMC, kn_s, kn_b, kb);
  v_gather<<<NBATCH * NHEAD * SEQ * HD / 256, 256, 0, stream>>>(qkvf, vb,
      NBATCH * NHEAD * SEQ * HD);

  // 4) flash attention: grid = (query blocks, heads, batch)
  flash_attn_wmma<<<dim3(SEQ / 64, NHEAD, NBATCH), 128, 0, stream>>>(qb, kb, vb, attn);

  // 5) output projection -> fp32 d_out
  gemm_f16_wmma<<<dim3(DIMC / 128, MROWS / 128), 256, 0, stream>>>(
      attn, wproj, proj_b, out, MROWS, DIMC, DIMC);
}
